// MHSA_49057116455367
// MI455X (gfx1250) — matrix operand order repaired and checked by reading, not compiled, not run on hardware
//
#include <hip/hip_runtime.h>
#include <hip/hip_bf16.h>
#include <math.h>

// ---------------------------------------------------------------------------
// MHSA (transposed channel attention) + proj + dwconv-gelu-dwconv, gfx1250.
// GEMMs: v_wmma_f32_16x16x32_bf16.  q/k streaming: TDM tensor_load_to_lds.
// ---------------------------------------------------------------------------

typedef __attribute__((ext_vector_type(16))) __bf16 v16bf;
typedef __attribute__((ext_vector_type(8)))  __bf16 v8bf;
typedef __attribute__((ext_vector_type(8)))  float  v8f;
typedef __attribute__((ext_vector_type(4)))  unsigned int v4u;
typedef __attribute__((ext_vector_type(8)))  int v8i;
typedef __attribute__((ext_vector_type(4)))  int v4i;

#define N_TOK 16384   // H*W = 128*128

static __device__ __forceinline__ v8f wmma_bf(v16bf a, v16bf b, v8f c) {
  return __builtin_amdgcn_wmma_f32_16x16x32_bf16(false, a, false, b,
                                                 (short)0, c, false, false);
}

// A-matrix fragment (16x32 bf16), source row-major (M, K) with K contiguous.
// Lane l: M = l&15 ; f[0..7] = K koff..koff+7, f[8..15] = K koff+16..koff+23,
// koff = 8*(l>>4).   (ISA 7.12.2, 16-bit A 16x32)
static __device__ __forceinline__ v16bf frag_a(const __bf16* p, int ldk) {
  const int lane = threadIdx.x & 31;
  const __bf16* r = p + (lane & 15) * ldk + ((lane >> 4) << 3);
  v8bf lo = *(const v8bf*)r;
  v8bf hi = *(const v8bf*)(r + 16);
  v16bf f;
#pragma unroll
  for (int i = 0; i < 8; ++i) { f[i] = lo[i]; f[i + 8] = hi[i]; }
  return f;
}

// B-matrix fragment (32x16 bf16), source is B^T row-major (N, K), K contiguous.
// Lane l: N = l&15 ; f[0..7] = K koff..koff+7, f[8..15] = K koff+16..koff+23, koff = 8*(l>>4).
static __device__ __forceinline__ v16bf frag_b(const __bf16* p, int ldk) {
  const int lane = threadIdx.x & 31;
  const __bf16* r = p + (lane & 15) * ldk + ((lane >> 4) << 3);
  v8bf lo = *(const v8bf*)r;
  v8bf hi = *(const v8bf*)(r + 16);
  v16bf f;
#pragma unroll
  for (int i = 0; i < 8; ++i) { f[i] = lo[i]; f[i + 8] = hi[i]; }
  return f;
}

// D tile store helper: lane l holds column N=l&15, rows M = 8*(l>>4)+r.
static __device__ __forceinline__ void store_row8(__bf16* p, v8f a) {
  v8bf o;
#pragma unroll
  for (int r = 0; r < 8; ++r) o[r] = (__bf16)a[r];
  *(v8bf*)p = o;
}

// LDS byte offset of a __shared__ object (LDS aperture keeps offset in [31:0]).
static __device__ __forceinline__ unsigned lds_off(const void* p) {
  return (unsigned)(unsigned long long)(size_t)p;
}

// TDM: async-load a 2D bf16 tile (64 rows x 128 elems, row stride 16384 elems)
// into LDS, inserting 16B of padding after every 256B row so the tile lands in
// a [64][136]-strided LDS buffer.  D# per CDNA5 ISA.
// amdgpu-toolchain builtin arity: (g0 v4u, g1 v8i, g2 v4i, g3 v4i, v8i, cpol).
static __device__ __forceinline__ void tdm_load_tile(unsigned lds_addr,
                                                     const __bf16* gptr) {
  unsigned long long ga = (unsigned long long)(size_t)gptr;
  v4u g0;
  g0[0] = 1u;                                    // count=1, user descriptor
  g0[1] = lds_addr;                              // lds_addr (bytes)
  g0[2] = (unsigned)ga;                          // global_addr[31:0]
  g0[3] = (unsigned)((ga >> 32) & 0x01ffffffu)   // global_addr[56:32]
          | (2u << 30);                          // type = 2 ("image")
  v8i g1;
  g1[0] = (int)((1u << 16)        // data_size = 1 -> 2 bytes
                | (1u << 20)      // pad_enable
                | (5u << 22)      // pad_interval: 2^5 8B-units = 256B
                | (3u << 25));    // pad_amount: 4 dwords = 16B
  g1[1] = (int)(16384u << 16);                   // tensor_dim0 = 16384 (lo16)
  g1[2] = (int)((16384u >> 16) | (64u << 16));   // dim0 hi | tensor_dim1 = 64
  g1[3] = (int)(128u << 16);                     // tile_dim0 = 128 elems
  g1[4] = 64;                                    // tile_dim1 = 64, tile_dim2=0
  g1[5] = 16384;                                 // tensor_dim0_stride lo32
  g1[6] = 0;                                     // stride hi | dim1_stride lo
  g1[7] = 0;                                     // dim1_stride hi
  v4i g2;
  g2[0] = 1;                                     // tensor_dim2 = 1 (benign)
  g2[1] = 1;                                     // tensor_dim3 = 1
  g2[2] = 0;                                     // tensor_dim2_stride lo
  g2[3] = 0;                                     // stride hi | tile_dim3 = 0
  v4i g3;
  g3[0] = 0;                                     // tensor_dim3_stride lo
  g3[1] = (int)(1u << 16);                       // stride hi | tensor_dim4 = 1
  g3[2] = 0;                                     // dim4 hi | tile_dim4 = 0
  g3[3] = 0;
  v8i g4 = {0, 0, 0, 0, 0, 0, 0, 0};             // unused 5th operand (VADDR4)
  __builtin_amdgcn_tensor_load_to_lds(g0, g1, g2, g3, g4, 0);
}

// --------------------------- kernel 0: prep --------------------------------
// WT[j][c]  = W^T for fused qkv  (j<256:q, <512:k, <768:v), bf16, K=c contig.
// WPt[o][j] = wp^T, bf16, K=j contiguous.  Also zeroes Sbuf (graph-replayable).
__global__ void __launch_bounds__(256) k_prep(
    const float* __restrict__ wq, const float* __restrict__ wk,
    const float* __restrict__ wv, const float* __restrict__ wp,
    __bf16* __restrict__ WT, __bf16* __restrict__ WPt,
    float* __restrict__ Sbuf) {
  int idx = blockIdx.x * 256 + threadIdx.x;          // 0..65535
  if (idx < 768 * 64) {
    int j = idx >> 6, c = idx & 63;
    const float* w = (j < 256) ? wq : ((j < 512) ? wk : wv);
    WT[idx] = (__bf16)w[c * 256 + (j & 255)];
  } else {
    int t = idx - 768 * 64;                          // 0..16383
    int o = t >> 8, jc = t & 255;
    WPt[t] = (__bf16)wp[jc * 64 + o];
  }
  Sbuf[idx] = 0.f;
  Sbuf[idx + 65536] = 0.f;                           // 131072 floats total
}

// ----------------------- kernel 1: LayerNorm + QKV --------------------------
// Block: 128 tokens. LN in fp32, GEMM (128x64)x(64x768) in bf16 WMMA.
// qT,kT layout: (b,h,d,N) N contiguous.  vB layout: (b,h,n,e) e contiguous.
__global__ void __launch_bounds__(256) k_ln_qkv(
    const float* __restrict__ x, const float* __restrict__ ln_w,
    const float* __restrict__ ln_b, const __bf16* __restrict__ WT,
    __bf16* __restrict__ qT, __bf16* __restrict__ kT,
    __bf16* __restrict__ vB) {
  __shared__ float  xs[128][65];     // fp32 staging, conflict-free stride
  __shared__ __align__(16) __bf16 xbf[128][72];  // normalized bf16
  const int tid = threadIdx.x;
  const int b  = blockIdx.x >> 7;
  const int n0 = (blockIdx.x & 127) << 7;
  const float* xp = x + ((size_t)b * 64) * N_TOK + n0;

  for (int idx = tid; idx < 64 * 128; idx += 256) {
    int c = idx >> 7, i = idx & 127;                 // coalesced over i
    xs[i][c] = xp[(size_t)c * N_TOK + i];
  }
  __syncthreads();
  if (tid < 128) {
    float s = 0.f, ss = 0.f;
#pragma unroll 8
    for (int c = 0; c < 64; ++c) { float v = xs[tid][c]; s += v; ss += v * v; }
    float mu = s * (1.f / 64.f);
    float var = ss * (1.f / 64.f) - mu * mu;
    float rs = rsqrtf(var + 1e-5f);
#pragma unroll 8
    for (int c = 0; c < 64; ++c)
      xbf[tid][c] = (__bf16)((xs[tid][c] - mu) * rs * ln_w[c] + ln_b[c]);
  }
  __syncthreads();

  const int w = tid >> 5, lane = tid & 31;
  const int mt = w;                                  // row tile 0..7
  v16bf a0 = frag_a(&xbf[mt * 16][0], 72);
  v16bf a1 = frag_a(&xbf[mt * 16][32], 72);
  const int b4 = b * 4;
  for (int jt = 0; jt < 48; ++jt) {
    const __bf16* wtp = WT + jt * 16 * 64;
    v16bf bb0 = frag_b(wtp, 64);
    v16bf bb1 = frag_b(wtp + 32, 64);
    v8f acc = {};
    acc = wmma_bf(a0, bb0, acc);
    acc = wmma_bf(a1, bb1, acc);
    int j = jt * 16 + (lane & 15);
    int nbase = n0 + mt * 16 + ((lane >> 4) << 3);
    if (j < 512) {                                   // q or k -> (b,h,d,N)
      __bf16* dst = (j < 256) ? qT : kT;
      int jj = j & 255;
      int h = jj >> 6, dd = jj & 63;
      store_row8(dst + (((size_t)(b4 + h) * 64 + dd) << 14) + nbase, acc);
    } else {                                         // v -> (b,h,n,e)
      int jj = j - 512;
      int h = jj >> 6, ee = jj & 63;
      size_t base = ((((size_t)(b4 + h)) << 14) + (size_t)nbase) * 64 + ee;
#pragma unroll
      for (int r = 0; r < 8; ++r) vB[base + (size_t)r * 64] = (__bf16)acc[r];
    }
  }
}

// ------------------- kernel 2a: S = q k^T  (K-split, TDM) -------------------
// grid = 32 (b,h) x 16 K-slices.  Wave 0 drives the Tensor Data Mover with
// double-buffered 64x128 q/k tiles (async, TENSORcnt); all 8 waves consume
// them with WMMA, then atomically accumulate into Sbuf[bh][64][64].
__global__ void __launch_bounds__(256) k_attn_qk(
    const __bf16* __restrict__ qT, const __bf16* __restrict__ kT,
    float* __restrict__ Sbuf) {
  __shared__ __align__(16) __bf16 qs[2][64][136];
  __shared__ __align__(16) __bf16 kss[2][64][136];
  const int tid = threadIdx.x, w = tid >> 5, lane = tid & 31;
  const int bh = blockIdx.x >> 4;
  const int kslice = blockIdx.x & 15;
  const __bf16* qp = qT + (((size_t)bh * 64) << 14);
  const __bf16* kp = kT + (((size_t)bh * 64) << 14);
  const int mt  = w >> 1;                    // tiles 2w,2w+1 share mt
  const int nt0 = (2 * w) & 3, nt1 = nt0 + 1;
  v8f acc0 = {}, acc1 = {};
  const int ck0 = kslice * 1024;

  if (w == 0) {                              // prologue: buffer 0 in flight
    tdm_load_tile(lds_off(&qs[0][0][0]),  qp + ck0);
    tdm_load_tile(lds_off(&kss[0][0][0]), kp + ck0);
  }
  for (int it = 0; it < 8; ++it) {
    const int cur = it & 1;
    if (w == 0) {
      if (it < 7) {                          // kick next pair, then wait cur
        const int nxt = cur ^ 1;
        tdm_load_tile(lds_off(&qs[nxt][0][0]),  qp + ck0 + (it + 1) * 128);
        tdm_load_tile(lds_off(&kss[nxt][0][0]), kp + ck0 + (it + 1) * 128);
        __builtin_amdgcn_s_wait_tensorcnt(2);   // in-order: cur pair done
      } else {
        __builtin_amdgcn_s_wait_tensorcnt(0);
      }
    }
    __syncthreads();                         // cur tiles visible to all waves
#pragma unroll
    for (int ksx = 0; ksx < 4; ++ksx) {
      v16bf a  = frag_a(&qs[cur][mt * 16][ksx * 32], 136);
      v16bf b0 = frag_b(&kss[cur][nt0 * 16][ksx * 32], 136);
      v16bf b1 = frag_b(&kss[cur][nt1 * 16][ksx * 32], 136);
      acc0 = wmma_bf(a, b0, acc0);
      acc1 = wmma_bf(a, b1, acc1);
    }
    __syncthreads();                         // done reading cur before reuse
  }
  float* sp = Sbuf + (size_t)bh * 4096;
  const int rr = (lane >> 4) << 3, cc = lane & 15;
#pragma unroll
  for (int r = 0; r < 8; ++r) {
    atomicAdd(&sp[(mt * 16 + rr + r) * 64 + nt0 * 16 + cc], acc0[r]);
    atomicAdd(&sp[(mt * 16 + rr + r) * 64 + nt1 * 16 + cc], acc1[r]);
  }
}

// ---------------------- kernel 2b: scale + softmax --------------------------
__global__ void k_softmax(const float* __restrict__ Sbuf,
                          const float* __restrict__ scale,
                          __bf16* __restrict__ attnb) {
  const int bh = blockIdx.x, d = threadIdx.x;        // 64 threads
  const float sc = scale[bh & 3];
  const float* row = Sbuf + (size_t)bh * 4096 + d * 64;
  float m = -3.4e38f;
  for (int e = 0; e < 64; ++e) m = fmaxf(m, row[e] * sc);
  float s = 0.f;
  for (int e = 0; e < 64; ++e) s += expf(row[e] * sc - m);
  const float inv = 1.f / s;
  __bf16* orow = attnb + (size_t)bh * 4096 + d * 64;
  for (int e = 0; e < 64; ++e)
    orow[e] = (__bf16)(expf(row[e] * sc - m) * inv);
}

// ------------------- kernel 2c: out = attn * v  (N-split) -------------------
// ao layout: (b, n, h*64+d) j contiguous  -> A-operand of projection GEMM.
__global__ void __launch_bounds__(256) k_attn_v(
    const __bf16* __restrict__ attnb, const __bf16* __restrict__ vB,
    __bf16* __restrict__ ao) {
  const int tid = threadIdx.x, w = tid >> 5, lane = tid & 31;
  const int bh = blockIdx.x >> 5, slice = blockIdx.x & 31;
  const int b = bh >> 2, h = bh & 3;
  const int mt = w & 3;
  const __bf16* apA = attnb + (size_t)bh * 4096 + mt * 16 * 64;
  v16bf af0 = frag_a(apA, 64);
  v16bf af1 = frag_a(apA + 32, 64);
  const __bf16* vp = vB + (((size_t)bh) << 14) * 64;
  __bf16* aop = ao + (((size_t)b) << 14) * 256 + h * 64;
  const int ntb = slice * 32;
  for (int nt = ntb + (w >> 2); nt < ntb + 32; nt += 2) {
    const __bf16* bpv = vp + (size_t)nt * 16 * 64;   // rows n, e contiguous
    v16bf b0 = frag_b(bpv, 64);
    v16bf b1 = frag_b(bpv + 32, 64);
    v8f acc = {};
    acc = wmma_bf(af0, b0, acc);
    acc = wmma_bf(af1, b1, acc);
    int n  = nt * 16 + (lane & 15);
    int dd = mt * 16 + ((lane >> 4) << 3);
    store_row8(aop + (size_t)n * 256 + dd, acc);
  }
}

// ----------------------- kernel 3: projection + bias ------------------------
__global__ void __launch_bounds__(256) k_proj(
    const __bf16* __restrict__ ao, const __bf16* __restrict__ WPt,
    const float* __restrict__ bp, float* __restrict__ proj) {
  const int tid = threadIdx.x, w = tid >> 5, lane = tid & 31;
  const size_t t0 = (size_t)blockIdx.x * 128;
  const int mt = w;
  const __bf16* apr = ao + (t0 + mt * 16) * 256;
  v16bf a[8];
#pragma unroll
  for (int ksx = 0; ksx < 8; ++ksx) a[ksx] = frag_a(apr + ksx * 32, 256);
#pragma unroll
  for (int ct = 0; ct < 4; ++ct) {
    const __bf16* bw = WPt + ct * 16 * 256;
    v8f acc = {};
#pragma unroll
    for (int ksx = 0; ksx < 8; ++ksx)
      acc = wmma_bf(a[ksx], frag_b(bw + ksx * 32, 256), acc);
    int o = ct * 16 + (lane & 15);
    float bias = bp[o];
    size_t row0 = t0 + mt * 16 + ((lane >> 4) << 3);
#pragma unroll
    for (int r = 0; r < 8; ++r) proj[(row0 + r) * 64 + o] = acc[r] + bias;
  }
}

// ------------- kernel 4: dwconv3x3 -> exact GELU -> dwconv3x3 ---------------
// proj reinterpreted (raw) as (B,64,128,128). 32x32 tile + halo 2 in LDS.
__global__ void __launch_bounds__(256) k_conv(
    const float* __restrict__ proj, const float* __restrict__ w1,
    const float* __restrict__ w2, float* __restrict__ out) {
  __shared__ float sin_[36][37];
  __shared__ float st1[34][37];
  const int tid = threadIdx.x;
  const int bc = blockIdx.x;                         // b*64 + c
  const int c  = bc & 63;
  const int y0 = blockIdx.y * 32, x0 = blockIdx.z * 32;
  const float* inp = proj + (size_t)bc * 16384;
  for (int idx = tid; idx < 36 * 36; idx += 256) {
    int i = idx / 36, j = idx % 36;
    int gy = y0 - 2 + i, gx = x0 - 2 + j;
    sin_[i][j] = (gy >= 0 && gy < 128 && gx >= 0 && gx < 128)
                     ? inp[gy * 128 + gx] : 0.f;
  }
  float k1[9], k2[9];
#pragma unroll
  for (int t = 0; t < 9; ++t) { k1[t] = w1[c * 9 + t]; k2[t] = w2[c * 9 + t]; }
  __syncthreads();
  for (int idx = tid; idx < 34 * 34; idx += 256) {
    int i = idx / 34, j = idx % 34;
    int gy = y0 - 1 + i, gx = x0 - 1 + j;
    float s = 0.f;
#pragma unroll
    for (int dy = 0; dy < 3; ++dy)
#pragma unroll
      for (int dx = 0; dx < 3; ++dx)
        s += k1[dy * 3 + dx] * sin_[i + dy][j + dx];
    float g = 0.5f * s * (1.f + erff(s * 0.70710678118654752f));
    st1[i][j] = (gy >= 0 && gy < 128 && gx >= 0 && gx < 128) ? g : 0.f;
  }
  __syncthreads();
  for (int idx = tid; idx < 32 * 32; idx += 256) {
    int i = idx >> 5, j = idx & 31;
    float s = 0.f;
#pragma unroll
    for (int dy = 0; dy < 3; ++dy)
#pragma unroll
      for (int dx = 0; dx < 3; ++dx)
        s += k2[dy * 3 + dx] * st1[i + dy][j + dx];
    out[(size_t)bc * 16384 + (y0 + i) * 128 + (x0 + j)] = s;
  }
}

// ---------------------------------------------------------------------------
extern "C" void kernel_launch(void* const* d_in, const int* in_sizes, int n_in,
                              void* d_out, int out_size, void* d_ws,
                              size_t ws_size, hipStream_t stream) {
  (void)in_sizes; (void)n_in; (void)out_size; (void)ws_size;
  const float* x     = (const float*)d_in[0];
  const float* ln_w  = (const float*)d_in[1];
  const float* ln_b  = (const float*)d_in[2];
  const float* wq    = (const float*)d_in[3];
  const float* wk    = (const float*)d_in[4];
  const float* wv    = (const float*)d_in[5];
  const float* scale = (const float*)d_in[6];
  const float* wp    = (const float*)d_in[7];
  const float* bp    = (const float*)d_in[8];
  const float* c1w   = (const float*)d_in[9];
  const float* c2w   = (const float*)d_in[10];
  float* out = (float*)d_out;

  // Workspace map (bytes):
  //   0        WT    (768x64 bf16)            98304
  //   98304    WPt   (64x256 bf16)            32768
  //   131072   Sbuf  (32x64x64 f32)          524288
  //   655360   attnb (32x64x64 bf16)         262144
  //   917504   qT    (8x4x64x16384 bf16)   67108864
  //   +qsz     kT                          67108864
  //   +qsz     vB                          67108864
  //   +qsz     ao                          67108864   total ~269.4 MB
  // proj (fp32 conv input, 33.5 MB) aliases qT, which is dead by then.
  char* ws = (char*)d_ws;
  __bf16* WT    = (__bf16*)(ws);
  __bf16* WPt   = (__bf16*)(ws + 98304);
  float*  Sbuf  = (float*)(ws + 131072);
  __bf16* attnb = (__bf16*)(ws + 655360);
  __bf16* qT    = (__bf16*)(ws + 917504);
  __bf16* kT    = qT + 33554432;
  __bf16* vB    = kT + 33554432;
  __bf16* ao    = vB + 33554432;
  float*  proj  = (float*)qT;                        // alias (q dead)

  k_prep   <<<256, 256, 0, stream>>>(wq, wk, wv, wp, WT, WPt, Sbuf);
  k_ln_qkv <<<1024, 256, 0, stream>>>(x, ln_w, ln_b, WT, qT, kT, vB);
  k_attn_qk<<<512, 256, 0, stream>>>(qT, kT, Sbuf);
  k_softmax<<<32, 64, 0, stream>>>(Sbuf, scale, attnb);
  k_attn_v <<<1024, 256, 0, stream>>>(attnb, vB, ao);
  k_proj   <<<1024, 256, 0, stream>>>(ao, WPt, bp, proj);
  k_conv   <<<dim3(512, 4, 4), 256, 0, stream>>>(proj, c1w, c2w, out);
}
